// LSTMAutoencoder_49658411876497
// MI455X (gfx1250) — compile-verified, matrix operand order checked by reading, not run on hardware
//
#include <hip/hip_runtime.h>
#include <hip/hip_bf16.h>

// ---------------------------------------------------------------------------
// LSTM autoencoder for MI455X (gfx1250, wave32).
// bf16 WMMA (v_wmma_f32_16x16x32_bf16) everywhere, f32 accumulate/gate math.
// Recurrent kernel: concat A-operand [x_t | h_{t-1}] staged in LDS, weights
// pre-packed into WMMA fragment order (one streaming pointer, 4 gate frags at
// immediate offsets), cell state in registers, 32 waves per direction,
// workgroup barrier per timestep. fwd/bwd directions run as 2 blocks.
// Gate nonlinearities use hardware v_rcp_f32 / v_exp_f32 (no IEEE divide).
// ---------------------------------------------------------------------------

#define DEVFN __device__ __forceinline__

constexpr int Bb  = 64;
constexpr int Tt  = 256;
constexpr int Pp  = 128;
constexpr int Hh  = 512;
constexpr int LAT = 128;

typedef __attribute__((ext_vector_type(16))) __bf16 v16bf;
typedef __attribute__((ext_vector_type(8)))  __bf16 v8bf;
typedef __attribute__((ext_vector_type(8)))  float  v8f;

DEVFN v8f zero8() {
  v8f z;
#pragma unroll
  for (int i = 0; i < 8; ++i) z[i] = 0.0f;
  return z;
}

// Load a 16x32 bf16 fragment from a row-major matrix. Lane l (half=l>>4,
// r=l&15) holds K elements {half*8..half*8+7} and {16+half*8..16+half*8+7}
// of row (row0+r): two 16-byte vector loads.
DEVFN v16bf load_tile(const __bf16* __restrict__ base, int ld, int row0, int k0) {
  const int lane = threadIdx.x & 31;
  const int half = lane >> 4;
  const int r    = row0 + (lane & 15);
  const __bf16* p = base + (long)r * ld + k0 + half * 8;
  union { v16bf v; v8bf h[2]; } u;
  u.h[0] = *(const v8bf*)(p);
  u.h[1] = *(const v8bf*)(p + 16);
  return u.v;
}

// Load a fragment that was pre-packed in per-lane order (16 contiguous bf16).
DEVFN v16bf load_packed(const __bf16* __restrict__ p) {
  union { v16bf v; v8bf h[2]; } u;
  u.h[0] = *(const v8bf*)(p);
  u.h[1] = *(const v8bf*)(p + 8);
  return u.v;
}

DEVFN v8f wmma_bf16(v16bf a, v16bf b, v8f c) {
  return __builtin_amdgcn_wmma_f32_16x16x32_bf16(
      /*neg_a=*/false, a, /*neg_b=*/false, b,
      /*c_mod=*/(short)0, c, /*reuse_a=*/false, /*reuse_b=*/false);
}

// Fast hardware reciprocal (v_rcp_f32, ~1ulp) -- avoids IEEE divide expansion.
DEVFN float fast_rcp(float x) { return __builtin_amdgcn_rcpf(x); }

DEVFN float sigm(float x) { return fast_rcp(1.0f + __expf(-x)); }
DEVFN float tanh_fast(float x) {
  x = fminf(15.0f, fmaxf(-15.0f, x));
  float e = __expf(2.0f * x);
  return (e - 1.0f) * fast_rcp(e + 1.0f);
}

// ---------------------------------------------------------------------------
// Elementwise helpers
// ---------------------------------------------------------------------------
__global__ void cast_f32_to_bf16(const float* __restrict__ in,
                                 __bf16* __restrict__ out, int n) {
  int i = blockIdx.x * blockDim.x + threadIdx.x;
  if (i < n) out[i] = (__bf16)in[i];
}

// xs[b,0,:] = start_token ; xs[b,t,:] = x[b,t-1,:]   (teacher forcing, ratio=1)
__global__ void build_xs_kernel(const float* __restrict__ x,
                                const float* __restrict__ start,
                                __bf16* __restrict__ xs, int n) {
  int i = blockIdx.x * blockDim.x + threadIdx.x;
  if (i >= n) return;
  int p = i % Pp;
  int t = (i / Pp) % Tt;
  xs[i] = (__bf16)(t == 0 ? start[p] : x[i - Pp]);
}

// ---------------------------------------------------------------------------
// Repack W = [Wih | Whh] (both (4H, *) f32 row-major) into WMMA fragment
// stream order, bf16:
//   out[((jb*nkb + kb)*4 + g)*512 + lane*16 + e]
//     = W[g*H + jb*16 + (lane&15)][ kb*32 + kmap(lane, e) ]
// where kmap = (e<8) ? (lane>>4)*8 + e : 16 + (lane>>4)*8 + (e-8).
// Column kk < Kin reads Wih, else Whh at (kk - Kin).
// ---------------------------------------------------------------------------
__global__ void pack_weights_kernel(const float* __restrict__ Wih,
                                    const float* __restrict__ Whh,
                                    int Kin, int nkb,
                                    __bf16* __restrict__ out, int n) {
  int i = blockIdx.x * blockDim.x + threadIdx.x;
  if (i >= n) return;
  const int e    = i & 15;
  const int lane = (i >> 4) & 31;
  const int g    = (i >> 9) & 3;
  const int rest = i >> 11;
  const int kb   = rest % nkb;
  const int jb   = rest / nkb;
  const int row  = g * Hh + jb * 16 + (lane & 15);
  const int kk   = kb * 32 +
                   ((e < 8) ? ((lane >> 4) * 8 + e) : (16 + (lane >> 4) * 8 + (e - 8)));
  float v = (kk < Kin) ? Wih[(long)row * Kin + kk]
                       : Whh[(long)row * Hh + (kk - Kin)];
  out[i] = (__bf16)v;
}

// ---------------------------------------------------------------------------
// Recurrent LSTM kernel. One 1024-thread block (32 waves) per direction;
// grid.x==2 runs two directions at once. Wave w owns h-columns [w*16,w*16+16).
// Per step: 4 batch tiles x 4 gate tiles, single K loop over Ktot = Kin + H
// reading the LDS-staged concat [x_t | h_{t-1}] and the packed weight stream.
//
// Dynamic LDS: astage (B x Ktot bf16) | hscratch (B x H bf16)
// ---------------------------------------------------------------------------
__global__ __launch_bounds__(1024) void lstm_kernel(
    const __bf16* __restrict__ xin, int xstride, int Kin, int Ktot, int T,
    const __bf16* __restrict__ Wp0, const float* __restrict__ bias0,
    const __bf16* __restrict__ Wp1, const float* __restrict__ bias1,
    int rev0, int rev1,
    const float* __restrict__ h0,                     // (B,H) f32 or null->0
    __bf16* __restrict__ out_h, int out_stride, int coloff0, int coloff1,
    __bf16* __restrict__ fin_h, int fin_stride, int fincol0, int fincol1) {
  const __bf16* Wp   = blockIdx.x ? Wp1   : Wp0;
  const float*  bias = blockIdx.x ? bias1 : bias0;
  const int rev    = blockIdx.x ? rev1    : rev0;
  const int coloff = blockIdx.x ? coloff1 : coloff0;
  const int fincol = blockIdx.x ? fincol1 : fincol0;

  extern __shared__ __align__(16) char smem_raw[];
  __bf16* astage   = (__bf16*)smem_raw;        // (B, Ktot) = [x_t | h_{t-1}]
  __bf16* hscratch = astage + Bb * Ktot;       // (B, H) h_t landing zone

  const int tid  = threadIdx.x;
  const int wave = tid >> 5;
  const int lane = tid & 31;
  const int half = lane >> 4;
  const int ln   = lane & 15;
  const int j0   = wave * 16;     // h-column block owned by this wave
  const int nkb  = Ktot >> 5;     // K blocks of 32

  float c[4][8];  // cell state: c[mt][v] at (row = mt*16+v+8*half, col = j0+ln)
#pragma unroll
  for (int mt = 0; mt < 4; ++mt)
#pragma unroll
    for (int v = 0; v < 8; ++v) c[mt][v] = 0.0f;

  float bi[4];  // per-gate bias for this lane's column (constant over time)
#pragma unroll
  for (int g = 0; g < 4; ++g) bi[g] = bias[g * Hh + j0 + ln];

  const int nxchunk = (Bb * Kin) >> 3;  // 16B chunks in the x_t row block
  const int nhchunk = (Bb * Hh) >> 3;   // 16B chunks in the h block

  for (int s = 0; s < T; ++s) {
    const int t = rev ? (T - 1 - s) : s;

    // ---- stage A = [x_t | h_{t-1}] into LDS (16B vector copies) ----
#pragma unroll 1
    for (int i = tid; i < nxchunk; i += 1024) {
      const int e  = i << 3;
      const int b  = e / Kin;
      const int kk = e - b * Kin;
      *(v8bf*)(astage + (long)b * Ktot + kk) =
          *(const v8bf*)(xin + ((long)b * T + t) * xstride + kk);
    }
    if (s == 0) {
      if (h0) {
#pragma unroll 1
        for (int i = tid; i < Bb * Hh; i += 1024)
          astage[(long)(i >> 9) * Ktot + Kin + (i & (Hh - 1))] = (__bf16)h0[i];
      } else {
#pragma unroll 1
        for (int i = tid; i < Bb * Hh; i += 1024)
          astage[(long)(i >> 9) * Ktot + Kin + (i & (Hh - 1))] = (__bf16)0.0f;
      }
    } else {
#pragma unroll 1
      for (int i = tid; i < nhchunk; i += 1024) {
        const int e   = i << 3;
        const int b   = e >> 9;         // e / H
        const int col = e & (Hh - 1);
        *(v8bf*)(astage + (long)b * Ktot + Kin + col) =
            *(const v8bf*)(hscratch + e);
      }
    }
    __syncthreads();  // A-stage ready; hscratch free to overwrite

    // ---- 4 batch tiles x 4 gate tiles, single fused K loop ----
#pragma unroll
    for (int mt = 0; mt < 4; ++mt) {
      const int m0 = mt * 16;
      v8f acc[4];
#pragma unroll
      for (int g = 0; g < 4; ++g) acc[g] = zero8();

      // single weight stream pointer; gate fragments at immediate offsets
      const __bf16* wp = Wp + ((long)wave * nkb) * 2048 + lane * 16;
#pragma unroll 1
      for (int kb = 0; kb < nkb; ++kb) {
        v16bf a = load_tile(astage, Ktot, m0, kb * 32);
#pragma unroll
        for (int g = 0; g < 4; ++g)
          acc[g] = wmma_bf16(a, load_packed(wp + g * 512), acc[g]);
        wp += 2048;
      }

      // gates (PyTorch order i,f,g,o); h_t -> hscratch (+ optional global)
#pragma unroll
      for (int v = 0; v < 8; ++v) {
        const int m = m0 + v + 8 * half;  // batch row
        float gi = acc[0][v] + bi[0];
        float gf = acc[1][v] + bi[1];
        float gg = acc[2][v] + bi[2];
        float go = acc[3][v] + bi[3];
        float cn = sigm(gf) * c[mt][v] + sigm(gi) * tanh_fast(gg);
        c[mt][v] = cn;
        float hn = sigm(go) * tanh_fast(cn);
        hscratch[m * Hh + j0 + ln] = (__bf16)hn;
        if (out_h)
          out_h[((long)m * T + t) * out_stride + coloff + j0 + ln] = (__bf16)hn;
      }
    }
    __syncthreads();  // h_t complete; astage free for next step
  }

  if (fin_h) {  // hscratch holds h_T
    for (int i = tid; i < Bb * Hh; i += 1024) {
      const int b   = i >> 9;
      const int col = i & (Hh - 1);
      fin_h[b * fin_stride + fincol + col] = hscratch[i];
    }
  }
}

// ---------------------------------------------------------------------------
// Generic WMMA GEMM: C[M,N] = act(A[M,K] @ W[N,K]^T + bias). One 16x16 C tile
// per wave, 8 waves per block. Optional f32 and/or bf16 outputs.
// ---------------------------------------------------------------------------
__global__ __launch_bounds__(256) void wmma_gemm_kernel(
    const __bf16* __restrict__ A, int lda,
    const __bf16* __restrict__ W, int ldw,
    const float* __restrict__ bias,
    int N, int K,
    float* __restrict__ Cf, __bf16* __restrict__ Cb, int ldc,
    int act, int tiles) {
  const int wid = blockIdx.x * (blockDim.x >> 5) + (threadIdx.x >> 5);
  if (wid >= tiles) return;  // wave-uniform: EXEC stays all-ones for WMMA
  const int ntn = N >> 4;
  const int mt = wid / ntn;
  const int nt = wid % ntn;
  const int lane = threadIdx.x & 31;
  const int half = lane >> 4;
  const int ln   = lane & 15;

  v8f acc = zero8();
#pragma unroll 1
  for (int k = 0; k < K; k += 32) {
    v16bf a = load_tile(A, lda, mt * 16, k);
    v16bf w = load_tile(W, ldw, nt * 16, k);
    acc = wmma_bf16(a, w, acc);
  }

  const int col = nt * 16 + ln;
  const float b = bias ? bias[col] : 0.0f;
#pragma unroll
  for (int v = 0; v < 8; ++v) {
    const int row = mt * 16 + v + 8 * half;
    float val = acc[v] + b;
    if (act == 1) val = sigm(val);
    if (Cf) Cf[(long)row * ldc + col] = val;
    if (Cb) Cb[(long)row * ldc + col] = (__bf16)val;
  }
}

// ---------------------------------------------------------------------------
// Host orchestration
// ---------------------------------------------------------------------------
extern "C" void kernel_launch(void* const* d_in, const int* in_sizes, int n_in,
                              void* d_out, int out_size, void* d_ws, size_t ws_size,
                              hipStream_t stream) {
  (void)in_sizes; (void)n_in; (void)out_size; (void)ws_size;

  const float* x        = (const float*)d_in[0];
  const float* e0f_Wih  = (const float*)d_in[1];
  const float* e0f_Whh  = (const float*)d_in[2];
  const float* e0f_b    = (const float*)d_in[3];
  const float* e0b_Wih  = (const float*)d_in[4];
  const float* e0b_Whh  = (const float*)d_in[5];
  const float* e0b_b    = (const float*)d_in[6];
  const float* e1f_Wih  = (const float*)d_in[7];
  const float* e1f_Whh  = (const float*)d_in[8];
  const float* e1f_b    = (const float*)d_in[9];
  const float* e1b_Wih  = (const float*)d_in[10];
  const float* e1b_Whh  = (const float*)d_in[11];
  const float* e1b_b    = (const float*)d_in[12];
  const float* enc_fc_W = (const float*)d_in[13];
  const float* enc_fc_b = (const float*)d_in[14];
  const float* dec_fc_W = (const float*)d_in[15];
  const float* dec_fc_b = (const float*)d_in[16];
  const float* start_tk = (const float*)d_in[17];
  const float* d0_Wih   = (const float*)d_in[18];
  const float* d0_Whh   = (const float*)d_in[19];
  const float* d0_b     = (const float*)d_in[20];
  const float* d1_Wih   = (const float*)d_in[21];
  const float* d1_Whh   = (const float*)d_in[22];
  const float* d1_b     = (const float*)d_in[23];
  const float* out_W    = (const float*)d_in[24];
  const float* out_b    = (const float*)d_in[25];
  // d_in[26] = teacher_forcing_ratio (== 1, teacher forcing always on)

  float* out_xhat = (float*)d_out;                         // (B,T,P)
  float* out_z    = out_xhat + (size_t)Bb * Tt * Pp;       // (B,LAT)

  // -- workspace layout (256B aligned) --
  size_t off = 0;
  auto alloc = [&](size_t bytes) -> char* {
    char* p = (char*)d_ws + off;
    off = (off + bytes + 255) & ~(size_t)255;
    return p;
  };
  auto allocB = [&](size_t n) { return (__bf16*)alloc(n * sizeof(__bf16)); };
  auto allocF = [&](size_t n) { return (float*)alloc(n * sizeof(float)); };

  __bf16* x_bf   = allocB((size_t)Bb * Tt * Pp);
  __bf16* xs_bf  = allocB((size_t)Bb * Tt * Pp);
  __bf16* l0_bf  = allocB((size_t)Bb * Tt * 2 * Hh);  // concat(f0, r0)
  __bf16* o0_bf  = allocB((size_t)Bb * Tt * Hh);
  __bf16* o1_bf  = allocB((size_t)Bb * Tt * Hh);

  // packed recurrent weights: 4H rows x Ktot cols, fragment order
  const int K_e0 = Pp + Hh;        // 640
  const int K_e1 = 2 * Hh + Hh;    // 1536
  const int K_d0 = Pp + Hh;        // 640
  const int K_d1 = Hh + Hh;        // 1024
  __bf16* wp_e0f = allocB((size_t)4 * Hh * K_e0);
  __bf16* wp_e0b = allocB((size_t)4 * Hh * K_e0);
  __bf16* wp_e1f = allocB((size_t)4 * Hh * K_e1);
  __bf16* wp_e1b = allocB((size_t)4 * Hh * K_e1);
  __bf16* wp_d0  = allocB((size_t)4 * Hh * K_d0);
  __bf16* wp_d1  = allocB((size_t)4 * Hh * K_d1);

  __bf16* wenc = allocB(LAT * 2 * Hh);
  __bf16* wdec = allocB(Hh * LAT);
  __bf16* wout = allocB(Pp * Hh);

  __bf16* encfin_bf = allocB((size_t)Bb * 2 * Hh);  // concat(h_fwd, h_bwd)
  __bf16* z_bf      = allocB((size_t)Bb * LAT);
  float*  h0d       = allocF((size_t)Bb * Hh);      // decoder initial h (f32)

  // -- 1) casts + weight packing --
  struct CJob { const float* s; __bf16* d; int n; };
  const CJob cjobs[] = {
      {x,        x_bf, Bb * Tt * Pp},
      {enc_fc_W, wenc, LAT * 2 * Hh},
      {dec_fc_W, wdec, Hh * LAT},
      {out_W,    wout, Pp * Hh},
  };
  for (const CJob& j : cjobs)
    cast_f32_to_bf16<<<(j.n + 255) / 256, 256, 0, stream>>>(j.s, j.d, j.n);

  struct PJob { const float* ih; const float* hh; int Kin; int Ktot; __bf16* d; };
  const PJob pjobs[] = {
      {e0f_Wih, e0f_Whh, Pp,     K_e0, wp_e0f},
      {e0b_Wih, e0b_Whh, Pp,     K_e0, wp_e0b},
      {e1f_Wih, e1f_Whh, 2 * Hh, K_e1, wp_e1f},
      {e1b_Wih, e1b_Whh, 2 * Hh, K_e1, wp_e1b},
      {d0_Wih,  d0_Whh,  Pp,     K_d0, wp_d0},
      {d1_Wih,  d1_Whh,  Hh,     K_d1, wp_d1},
  };
  for (const PJob& j : pjobs) {
    const int n = 4 * Hh * j.Ktot;
    pack_weights_kernel<<<(n + 255) / 256, 256, 0, stream>>>(
        j.ih, j.hh, j.Kin, j.Ktot >> 5, j.d, n);
  }

  build_xs_kernel<<<(Bb * Tt * Pp + 255) / 256, 256, 0, stream>>>(
      x, start_tk, xs_bf, Bb * Tt * Pp);

  // dynamic LDS: astage (B x Ktot) + hscratch (B x H), bf16
  auto lds_bytes = [](int Ktot) {
    return (size_t)(Bb * Ktot + Bb * Hh) * sizeof(__bf16);
  };

  // -- 2) encoder layer 0 (bidirectional): writes l0 = concat(f0, r0) --
  lstm_kernel<<<2, 1024, lds_bytes(K_e0), stream>>>(
      x_bf, Pp, Pp, K_e0, Tt,
      wp_e0f, e0f_b, wp_e0b, e0b_b,
      /*rev0=*/0, /*rev1=*/1, /*h0=*/nullptr,
      l0_bf, 2 * Hh, /*coloff0=*/0, /*coloff1=*/Hh,
      /*fin_h=*/nullptr, 0, 0, 0);

  // -- 3) encoder layer 1 (bidirectional): only final h needed --
  lstm_kernel<<<2, 1024, lds_bytes(K_e1), stream>>>(
      l0_bf, 2 * Hh, 2 * Hh, K_e1, Tt,
      wp_e1f, e1f_b, wp_e1b, e1b_b,
      /*rev0=*/0, /*rev1=*/1, /*h0=*/nullptr,
      /*out_h=*/nullptr, 0, 0, 0,
      encfin_bf, 2 * Hh, /*fincol0=*/0, /*fincol1=*/Hh);

  // -- 4) z = encfin @ enc_fc_W^T + b  -> d_out(z) f32 + bf16 for dec_fc --
  {
    int tiles = (Bb / 16) * (LAT / 16);  // 32
    wmma_gemm_kernel<<<(tiles + 7) / 8, 256, 0, stream>>>(
        encfin_bf, 2 * Hh, wenc, 2 * Hh, enc_fc_b,
        LAT, 2 * Hh, out_z, z_bf, LAT, /*act=*/0, tiles);
  }

  // -- 5) h0d = z @ dec_fc_W^T + b  (f32, h0 for both decoder layers) --
  {
    int tiles = (Bb / 16) * (Hh / 16);  // 128
    wmma_gemm_kernel<<<(tiles + 7) / 8, 256, 0, stream>>>(
        z_bf, LAT, wdec, LAT, dec_fc_b,
        Hh, LAT, h0d, nullptr, Hh, /*act=*/0, tiles);
  }

  // -- 6) decoder layer 0: o0 = LSTM(xs; h0=h0d, c0=0) --
  lstm_kernel<<<1, 1024, lds_bytes(K_d0), stream>>>(
      xs_bf, Pp, Pp, K_d0, Tt,
      wp_d0, d0_b, nullptr, nullptr,
      /*rev0=*/0, 0, h0d,
      o0_bf, Hh, 0, 0,
      nullptr, 0, 0, 0);

  // -- 7) decoder layer 1: o1 = LSTM(o0; h0=h0d, c0=0) --
  lstm_kernel<<<1, 1024, lds_bytes(K_d1), stream>>>(
      o0_bf, Hh, Hh, K_d1, Tt,
      wp_d1, d1_b, nullptr, nullptr,
      /*rev0=*/0, 0, h0d,
      o1_bf, Hh, 0, 0,
      nullptr, 0, 0, 0);

  // -- 8) x_hat = sigmoid(o1 @ out_W^T + b) -> d_out f32 --
  {
    int tiles = (Bb * Tt / 16) * (Pp / 16);  // 8192
    wmma_gemm_kernel<<<(tiles + 7) / 8, 256, 0, stream>>>(
        o1_bf, Hh, wout, Hh, out_b,
        Pp, Hh, out_xhat, nullptr, Pp, /*act=*/1, tiles);
  }
}
